// SSMLayer_20031727468878
// MI455X (gfx1250) — compile-verified, matrix operand order checked by reading, not run on hardware
//
#include <hip/hip_runtime.h>
#include <hip/hip_bf16.h>

// SSM layer: out[b,t] = (h_t = tanh(h_{t-1} @ A + x[b,t] @ Bm)) @ C
// B=32, T=2048, I=128, H=256.  All heavy math via v_wmma_f32_16x16x32_f16.
//
// Weights are repacked once into WMMA B-fragment-major layout so every
// fragment load is one contiguous 32-byte vector load (2x global_load_b128).

typedef __attribute__((ext_vector_type(16))) _Float16 v16h;
typedef __attribute__((ext_vector_type(8)))  float    v8f;

constexpr int B_SZ = 32;
constexpr int T    = 2048;
constexpr int KI   = 128;      // input dim I
constexpr int H    = 256;      // hidden dim
constexpr int M    = B_SZ * T; // 65536 rows for the big GEMMs

// ---------------------------------------------------------------------------
// fast tanh: gfx1250 has a TANH transcendental
// ---------------------------------------------------------------------------
__device__ __forceinline__ float fast_tanh(float x) {
#if __has_builtin(__builtin_amdgcn_tanhf)
  return __builtin_amdgcn_tanhf(x);
#else
  return tanhf(x);
#endif
}

// ---------------------------------------------------------------------------
// WMMA fragment helpers (ISA 7.12.2 layouts, wave32)
// A-frag (16x32 f16, MxK): lane L<16 holds row M=L with K = k0+{0..7,16..23},
//                          lane L>=16 holds row M=L-16, K = k0+{8..15,24..31}.
// p0 must point at &base[row*ld + k0 + ((lane>>4)<<3)]
// ---------------------------------------------------------------------------
__device__ __forceinline__ v16h frag_a_f16(const _Float16* __restrict__ p0) {
  v16h a;
#pragma unroll
  for (int e = 0; e < 8; ++e) a[e] = p0[e];
#pragma unroll
  for (int e = 0; e < 8; ++e) a[e + 8] = p0[e + 16];
  return a;
}

// same, but f32 source (read-once stream -> non-temporal) converted to f16
__device__ __forceinline__ v16h frag_a_f32cvt_nt(const float* __restrict__ p0) {
  v16h a;
#pragma unroll
  for (int e = 0; e < 8; ++e) a[e] = (_Float16)__builtin_nontemporal_load(p0 + e);
#pragma unroll
  for (int e = 0; e < 8; ++e)
    a[e + 8] = (_Float16)__builtin_nontemporal_load(p0 + e + 16);
  return a;
}

// Packed B-fragment: one contiguous v16h per (n-tile, k-step, lane).
__device__ __forceinline__ v16h frag_b_packed(const _Float16* __restrict__ base,
                                              int nt, int kk, int nkk) {
  const int lane = threadIdx.x & 31;
  const size_t off = ((((size_t)nt * nkk + kk) * 32) + lane) * 16;
  return *(const v16h*)(base + off);
}

__device__ __forceinline__ v8f wmma_f16(v16h a, v16h b, v8f c) {
  return __builtin_amdgcn_wmma_f32_16x16x32_f16(
      /*neg_a=*/false, a, /*neg_b=*/false, b,
      /*c_mod=*/(short)0, c, /*reuse_a=*/false, /*reuse_b=*/false);
}

// ---------------------------------------------------------------------------
// Kernel 0: pack f32 weight [K][N] row-major -> f16 B-fragment-major layout
// packed[(((nt*nkk)+kk)*32+lane)*16+e] = w[k][n],
//   k = kk*32 + 8*(lane>=16) + (e<8 ? e : e+8),  n = nt*16 + (lane&15)
// ---------------------------------------------------------------------------
__global__ void __launch_bounds__(256)
k_pack(const float* __restrict__ src, _Float16* __restrict__ dst, int K, int N) {
  const int nkk = K >> 5;
  int p = blockIdx.x * blockDim.x + threadIdx.x;
  if (p >= K * N) return;
  const int e    = p & 15;
  const int lane = (p >> 4) & 31;
  const int rem  = p >> 9;
  const int kk   = rem % nkk;
  const int nt   = rem / nkk;
  const int k    = kk * 32 + ((lane >> 4) << 3) + (e < 8 ? e : e + 8);
  const int n    = nt * 16 + (lane & 15);
  dst[p] = (_Float16)src[(size_t)k * N + n];
}

// ---------------------------------------------------------------------------
// Kernel 1: xB = x @ Bm   [M, 128] x [128, 256] -> f16 [M, 256]
// 8 waves/block; each wave owns a 16-row M-tile, loads its A-frags once,
// sweeps all 16 N-tiles with packed weight fragments (WGP$/L2 resident).
// ---------------------------------------------------------------------------
__global__ void __launch_bounds__(256)
k_proj(const float* __restrict__ x, const _Float16* __restrict__ BmP,
       _Float16* __restrict__ xBH) {
  const int lane = threadIdx.x & 31;
  const int wv   = threadIdx.x >> 5;
  const int m0   = (blockIdx.x * 8 + wv) * 16;
  const int koff = (lane >> 4) << 3;
  const int hi8  = koff;
  const int ncol = lane & 15;
  const int row  = m0 + (lane & 15);

  v16h af[KI / 32];
#pragma unroll
  for (int kk = 0; kk < KI / 32; ++kk)
    af[kk] = frag_a_f32cvt_nt(x + (size_t)row * KI + kk * 32 + koff);

#pragma unroll 4
  for (int nt = 0; nt < H / 16; ++nt) {
    v8f acc = {0.f, 0.f, 0.f, 0.f, 0.f, 0.f, 0.f, 0.f};
#pragma unroll
    for (int kk = 0; kk < KI / 32; ++kk)
      acc = wmma_f16(af[kk], frag_b_packed(BmP, nt, kk, KI / 32), acc);
#pragma unroll
    for (int r = 0; r < 8; ++r)
      xBH[(size_t)(m0 + r + hi8) * H + nt * 16 + ncol] = (_Float16)acc[r];
  }
}

// ---------------------------------------------------------------------------
// Kernel 2: sequential scan  h_t = tanh(h_{t-1} @ A + xB_t)
// 2 workgroups (16 batch rows each), 512 threads = 16 waves, one 16x16 N-tile
// per wave. Recurrence weight pinned in VGPR fragments. h double-buffered in
// LDS; one barrier per step. xB loads hoisted to the top of the step so the
// (L2-resident) global loads overlap the ds_load + WMMA dependency chain.
// ---------------------------------------------------------------------------
constexpr int HP = H + 8;  // padded LDS row

__global__ void __launch_bounds__(512)
k_scan(const _Float16* __restrict__ xBH, const _Float16* __restrict__ AP,
       _Float16* __restrict__ hsH) {
  __shared__ _Float16 hbuf[2][16][HP];

  const int lane = threadIdx.x & 31;
  const int wv   = threadIdx.x >> 5;   // 0..15 -> N-tile
  const int n0   = wv * 16;
  const int b0   = blockIdx.x * 16;
  const int koff = (lane >> 4) << 3;
  const int hi8  = koff;
  const int ncol = lane & 15;
  const int arow = lane & 15;

  // zero h0 buffer
  for (int i = threadIdx.x; i < 16 * HP; i += blockDim.x)
    (&hbuf[0][0][0])[i] = (_Float16)0.f;

  // preload the 8 K-fragments of the recurrence weight for this wave's N-tile
  v16h wfr[8];
#pragma unroll
  for (int kk = 0; kk < 8; ++kk) wfr[kk] = frag_b_packed(AP, wv, kk, 8);

  __syncthreads();

  for (int t = 0; t < T; ++t) {
    const int pb = t & 1;

    // issue this step's input-projection loads first (independent of h)
    float xbv[8];
#pragma unroll
    for (int r = 0; r < 8; ++r) {
      const size_t idx = ((size_t)(b0 + r + hi8) * T + t) * H + n0 + ncol;
      xbv[r] = (float)xBH[idx];
    }

    // load h_{t-1} A-fragments from LDS (contiguous 16B chunks -> ds_load_b128)
    v16h hfr[8];
#pragma unroll
    for (int kk = 0; kk < 8; ++kk)
      hfr[kk] = frag_a_f16(&hbuf[pb][arow][kk * 32 + koff]);

    v8f acc = {0.f, 0.f, 0.f, 0.f, 0.f, 0.f, 0.f, 0.f};
#pragma unroll
    for (int kk = 0; kk < 8; ++kk) acc = wmma_f16(hfr[kk], wfr[kk], acc);

    // h_t = tanh(h_{t-1}A + xB); write LDS double buffer + global hs
#pragma unroll
    for (int r = 0; r < 8; ++r) {
      const int rl = r + hi8;
      const size_t idx = ((size_t)(b0 + rl) * T + t) * H + n0 + ncol;
      _Float16 hh = (_Float16)fast_tanh(acc[r] + xbv[r]);
      hbuf[1 - pb][rl][n0 + ncol] = hh;
      hsH[idx] = hh;
    }
    __syncthreads();
  }
}

// ---------------------------------------------------------------------------
// Kernel 3: out = hs @ C   f16 [M, 256] x [256, 256] -> f32 [M, 256]
// Same wave-owns-M-tile structure as k_proj; write-once output goes out
// non-temporally so it doesn't evict L2-resident hs/xB.
// ---------------------------------------------------------------------------
__global__ void __launch_bounds__(256)
k_out(const _Float16* __restrict__ hsH, const _Float16* __restrict__ CP,
      float* __restrict__ out) {
  const int lane = threadIdx.x & 31;
  const int wv   = threadIdx.x >> 5;
  const int m0   = (blockIdx.x * 8 + wv) * 16;
  const int koff = (lane >> 4) << 3;
  const int hi8  = koff;
  const int ncol = lane & 15;
  const int row  = m0 + (lane & 15);

  v16h af[H / 32];
#pragma unroll
  for (int kk = 0; kk < H / 32; ++kk)
    af[kk] = frag_a_f16(hsH + (size_t)row * H + kk * 32 + koff);

#pragma unroll 4
  for (int nt = 0; nt < H / 16; ++nt) {
    v8f acc = {0.f, 0.f, 0.f, 0.f, 0.f, 0.f, 0.f, 0.f};
#pragma unroll
    for (int kk = 0; kk < H / 32; ++kk)
      acc = wmma_f16(af[kk], frag_b_packed(CP, nt, kk, H / 32), acc);
#pragma unroll
    for (int r = 0; r < 8; ++r)
      __builtin_nontemporal_store(
          acc[r], out + (size_t)(m0 + r + hi8) * H + nt * 16 + ncol);
  }
}

// ---------------------------------------------------------------------------
extern "C" void kernel_launch(void* const* d_in, const int* in_sizes, int n_in,
                              void* d_out, int out_size, void* d_ws,
                              size_t ws_size, hipStream_t stream) {
  const float* x  = (const float*)d_in[0];  // [32, 2048, 128]
  const float* A  = (const float*)d_in[1];  // [256, 256]
  const float* Bm = (const float*)d_in[2];  // [128, 256]
  const float* C  = (const float*)d_in[3];  // [256, 256]
  float* out = (float*)d_out;               // [32, 2048, 256]

  // workspace layout (all 32B-aligned):
  char* p = (char*)d_ws;
  _Float16* AP  = (_Float16*)p; p += (size_t)H * H * sizeof(_Float16);
  _Float16* BmP = (_Float16*)p; p += (size_t)KI * H * sizeof(_Float16);
  _Float16* CP  = (_Float16*)p; p += (size_t)H * H * sizeof(_Float16);
  _Float16* xBH = (_Float16*)p; p += (size_t)M * H * sizeof(_Float16);
  _Float16* hsH = (_Float16*)p;

  k_pack<<<(H * H + 255) / 256, 256, 0, stream>>>(A, AP, H, H);
  k_pack<<<(KI * H + 255) / 256, 256, 0, stream>>>(Bm, BmP, KI, H);
  k_pack<<<(H * H + 255) / 256, 256, 0, stream>>>(C, CP, H, H);

  k_proj<<<M / 128, 256, 0, stream>>>(x, BmP, xBH);
  k_scan<<<B_SZ / 16, 512, 0, stream>>>(xBH, AP, hsH);
  k_out<<<M / 128, 256, 0, stream>>>(hsH, CP, out);
}
